// LTXAttention_19524921328130
// MI455X (gfx1250) — compile-verified, matrix operand order checked by reading, not run on hardware
//
#include <hip/hip_runtime.h>
#include <hip/hip_bf16.h>
#include <stdint.h>

// ---------------------------------------------------------------------------
// LTX attention block for MI455X (gfx1250, wave32, WMMA).
// All matmuls (QKV proj, Q@K^T, P@V, O proj) run on V_WMMA_F32_16X16X32_BF16.
// LDS staging uses GLOBAL_LOAD_ASYNC_TO_LDS_B128 (ASYNCcnt) when available.
// ---------------------------------------------------------------------------

#define SEQ    2048
#define DMODEL 2048
#define NHEADS 32
#define HD     64

typedef __attribute__((ext_vector_type(16))) __bf16 bf16x16;
typedef __attribute__((ext_vector_type(8)))  __bf16 bf16x8;
typedef __attribute__((ext_vector_type(8)))  float  v8f;

#if defined(__AMDGCN__) && __has_builtin(__builtin_amdgcn_global_load_async_to_lds_b128) && \
    __has_builtin(__builtin_amdgcn_s_wait_asynccnt)
#define ASYNC_LDS 1
#else
#define ASYNC_LDS 0
#endif

#if ASYNC_LDS
typedef __attribute__((vector_size(16))) int i32x4_v;
typedef __attribute__((address_space(1))) i32x4_v* g_v4i_ptr;
typedef __attribute__((address_space(3))) i32x4_v* l_v4i_ptr;
#endif

// Copy 16B global -> LDS. Async path: no VGPR data, tracked by ASYNCcnt.
static __device__ __forceinline__ void cp16(__bf16* dst_lds, const __bf16* src_glob) {
#if ASYNC_LDS
    // LDS generic pointer low 32 bits == LDS byte offset (aperture truncation).
    __builtin_amdgcn_global_load_async_to_lds_b128(
        (g_v4i_ptr)(uintptr_t)src_glob,
        (l_v4i_ptr)(uint32_t)(uintptr_t)dst_lds,
        0, 0);
#else
    *(bf16x8*)dst_lds = *(const bf16x8*)src_glob;
#endif
}

static __device__ __forceinline__ void cp_commit() {
#if ASYNC_LDS
    __builtin_amdgcn_s_wait_asynccnt(0);
#endif
}

// Build a 16-element A/B fragment from two contiguous 8-element (16B) chunks.
// ISA 16-bit fragment layout: lane holds K = {off..off+7} and {off+16..off+23},
// off = 8 * (lane >= 16).  Callers pass p0 = base+off, p1 = base+off+16.
static __device__ __forceinline__ bf16x16 frag16(const __bf16* p0, const __bf16* p1) {
    bf16x8 a = *(const bf16x8*)p0;
    bf16x8 b = *(const bf16x8*)p1;
    return __builtin_shufflevector(a, b, 0,1,2,3,4,5,6,7,8,9,10,11,12,13,14,15);
}

static __device__ __forceinline__ v8f wmma_bf16(bf16x16 a, bf16x16 b, v8f c) {
    return __builtin_amdgcn_wmma_f32_16x16x32_bf16(false, a, false, b, (short)0, c,
                                                   false, false);
}

// ---------------------------------------------------------------------------
// 1) f32 -> bf16 row-major convert (hidden_states)
// ---------------------------------------------------------------------------
__global__ __launch_bounds__(256) void cvt_bf16_kernel(const float* __restrict__ x,
                                                       __bf16* __restrict__ y) {
    size_t i = ((size_t)blockIdx.x * 256 + threadIdx.x) * 8;
    float4 a = *(const float4*)(x + i);
    float4 b = *(const float4*)(x + i + 4);
    bf16x8 o;
    o[0] = (__bf16)a.x; o[1] = (__bf16)a.y; o[2] = (__bf16)a.z; o[3] = (__bf16)a.w;
    o[4] = (__bf16)b.x; o[5] = (__bf16)b.y; o[6] = (__bf16)b.z; o[7] = (__bf16)b.w;
    *(bf16x8*)(y + i) = o;
}

// ---------------------------------------------------------------------------
// 2) Weight transpose + convert: Wt[n][k] = (bf16)W[k][n]
// ---------------------------------------------------------------------------
__global__ __launch_bounds__(256) void transpose_w_kernel(const float* __restrict__ w,
                                                          __bf16* __restrict__ wt) {
    __shared__ float tile[32][33];
    const int n0 = blockIdx.x * 32, k0 = blockIdx.y * 32;
    const int tx = threadIdx.x, ty = threadIdx.y;
    for (int j = ty; j < 32; j += 8)
        tile[j][tx] = w[(size_t)(k0 + j) * DMODEL + n0 + tx];
    __syncthreads();
    for (int j = ty; j < 32; j += 8)
        wt[(size_t)(n0 + j) * DMODEL + k0 + tx] = (__bf16)tile[tx][j];
}

// ---------------------------------------------------------------------------
// 3) WMMA GEMM: C[M,N] = A[M,K](bf16,row) @ Bt[N,K](bf16) + bias, M=N=K=2048.
//    Block 128x128, 8 waves (4x2), 32x64 per wave (2x4 WMMA accumulators).
//    Double-buffered LDS; async global->LDS staging; 1 barrier / K-step.
// ---------------------------------------------------------------------------
#define BM  128
#define BN  128
#define BK  32
#define LDT 40   // padded LDS row stride (bf16 elems) -> 80B, 16B-aligned chunks

__global__ __launch_bounds__(256) void gemm_bf16_kernel(const __bf16* __restrict__ A,
                                                        const __bf16* __restrict__ Bt,
                                                        const float* __restrict__ bias,
                                                        float* __restrict__ C) {
    __shared__ __bf16 As[2][BM][LDT];
    __shared__ __bf16 Bs[2][BN][LDT];
    const int t    = threadIdx.x;
    const int lane = t & 31, wid = t >> 5;
    const int ln   = lane & 15, half = lane >> 4;
    const int mwave = wid >> 1;   // 0..3  (rows of waves)
    const int nwave = wid & 1;    // 0..1  (cols of waves)
    const int mbase = blockIdx.y * BM, nbase = blockIdx.x * BN;

    // 512 16B chunks per matrix per tile -> 2 per thread per matrix
    auto stage = [&](int kb, int buf) {
#pragma unroll
        for (int i = 0; i < 2; i++) {
            int idx = t + i * 256;
            int row = idx >> 2, c = (idx & 3) * 8;
            cp16(&As[buf][row][c], A  + (size_t)(mbase + row) * DMODEL + kb + c);
            cp16(&Bs[buf][row][c], Bt + (size_t)(nbase + row) * DMODEL + kb + c);
        }
    };

    v8f acc[2][4] = {};
    stage(0, 0);
    cp_commit();
    __syncthreads();

    const int NIT = DMODEL / BK;
    for (int i = 0; i < NIT; i++) {
        const int cur = i & 1;
        if (i + 1 < NIT) stage((i + 1) * BK, cur ^ 1);  // overlap with compute

        bf16x16 af[2], bfm[4];
#pragma unroll
        for (int mi = 0; mi < 2; mi++) {
            const __bf16* p = &As[cur][mwave * 32 + mi * 16 + ln][half * 8];
            af[mi] = frag16(p, p + 16);
        }
#pragma unroll
        for (int ni = 0; ni < 4; ni++) {
            const __bf16* p = &Bs[cur][nwave * 64 + ni * 16 + ln][half * 8];
            bfm[ni] = frag16(p, p + 16);
        }
#pragma unroll
        for (int mi = 0; mi < 2; mi++)
#pragma unroll
            for (int ni = 0; ni < 4; ni++)
                acc[mi][ni] = wmma_bf16(af[mi], bfm[ni], acc[mi][ni]);

        cp_commit();       // next tile resident in LDS
        __syncthreads();
    }

    // C/D layout: VGPR r, lane L -> (M = r + 8*(L>=16), N = L&15)
#pragma unroll
    for (int mi = 0; mi < 2; mi++)
#pragma unroll
        for (int ni = 0; ni < 4; ni++) {
            int n = nbase + nwave * 64 + ni * 16 + ln;
            float bb = bias[n];
#pragma unroll
            for (int r = 0; r < 8; r++) {
                int m = mbase + mwave * 32 + mi * 16 + r + half * 8;
                C[(size_t)m * DMODEL + n] = acc[mi][ni][r] + bb;
            }
        }
}

// ---------------------------------------------------------------------------
// 4) Fused RMSNorm (full D) + RoPE + repack:
//    Qb,Kb: [h][s][64] bf16;  Vt: [h][64][s] bf16
// ---------------------------------------------------------------------------
static __device__ __forceinline__ void qk_row(const float* __restrict__ X,
                                              const float* __restrict__ w,
                                              const float* __restrict__ cs,
                                              const float* __restrict__ sn,
                                              __bf16* __restrict__ out,
                                              int s, int t, float* red) {
    const int d = t * 8;
    const size_t off = (size_t)s * DMODEL + d;
    float4 a = *(const float4*)(X + off);
    float4 b = *(const float4*)(X + off + 4);
    float x[8] = {a.x, a.y, a.z, a.w, b.x, b.y, b.z, b.w};
    float ss = 0.f;
#pragma unroll
    for (int i = 0; i < 8; i++) ss += x[i] * x[i];
    red[t] = ss;
    __syncthreads();
    for (int st2 = 128; st2 > 0; st2 >>= 1) {
        if (t < st2) red[t] += red[t + st2];
        __syncthreads();
    }
    float inv = rsqrtf(red[0] / (float)DMODEL + 1e-5f);
    __syncthreads();

    bf16x8 o;
#pragma unroll
    for (int i = 0; i < 4; i++) {
        float xr = x[2 * i]     * inv * w[d + 2 * i];
        float xi = x[2 * i + 1] * inv * w[d + 2 * i + 1];
        float c0 = cs[off + 2 * i], c1 = cs[off + 2 * i + 1];
        float s0 = sn[off + 2 * i], s1 = sn[off + 2 * i + 1];
        o[2 * i]     = (__bf16)(xr * c0 - xi * s0);
        o[2 * i + 1] = (__bf16)(xi * c1 + xr * s1);
    }
    const int h = d >> 6, dd = d & 63;
    *(bf16x8*)(out + ((size_t)h * SEQ + s) * HD + dd) = o;
}

__global__ __launch_bounds__(256) void norm_rope_pack_kernel(
        const float* __restrict__ Qf, const float* __restrict__ Kf,
        const float* __restrict__ Vf,
        const float* __restrict__ nqw, const float* __restrict__ nkw,
        const float* __restrict__ cs,  const float* __restrict__ sn,
        __bf16* __restrict__ Qb, __bf16* __restrict__ Kb, __bf16* __restrict__ Vt) {
    __shared__ float red[256];
    const int s = blockIdx.x, t = threadIdx.x;
    qk_row(Qf, nqw, cs, sn, Qb, s, t, red);
    qk_row(Kf, nkw, cs, sn, Kb, s, t, red);
    const int d = t * 8;
    const size_t off = (size_t)s * DMODEL + d;
    float4 a = *(const float4*)(Vf + off);
    float4 b = *(const float4*)(Vf + off + 4);
    float x[8] = {a.x, a.y, a.z, a.w, b.x, b.y, b.z, b.w};
    const int h = d >> 6, dd = d & 63;
#pragma unroll
    for (int i = 0; i < 8; i++)
        Vt[((size_t)h * HD + dd + i) * SEQ + s] = (__bf16)x[i];
}

// ---------------------------------------------------------------------------
// 5) Flash attention: grid (S/64 q-blocks, 32 heads), 4 waves x 16 q-rows.
//    K/V tiles staged once per block in LDS (shared by all 4 waves, async
//    copies); online softmax with 16-lane shfl_xor row reductions; P routed
//    through a per-wave LDS tile to re-lay it out as a WMMA A-fragment.
// ---------------------------------------------------------------------------
__global__ __launch_bounds__(128) void attn_kernel(const __bf16* __restrict__ Qb,
                                                   const __bf16* __restrict__ Kb,
                                                   const __bf16* __restrict__ Vt,
                                                   __bf16* __restrict__ Ob) {
    __shared__ __bf16 Ks[32][72];        // [key][d]  pad: 144B stride
    __shared__ __bf16 Vs[64][40];        // [d][s]    pad: 80B stride
    __shared__ __bf16 pbuf[4][16][32];   // per-wave P relayout tile
    const int t = threadIdx.x, lane = t & 31, wid = t >> 5;
    const int ln = lane & 15, half = lane >> 4;
    const int h  = blockIdx.y;
    const int q0 = blockIdx.x * 64 + wid * 16;

    // Q fragments (persistent): A 16x32, head_dim split into two 32-chunks
    const __bf16* Qh = Qb + ((size_t)h * SEQ + q0 + ln) * HD;
    bf16x16 aq[2];
#pragma unroll
    for (int ks = 0; ks < 2; ks++) {
        const __bf16* p = Qh + ks * 32 + half * 8;
        aq[ks] = frag16(p, p + 16);
    }

    v8f acc[4] = {};
    float mI[8], lI[8];
#pragma unroll
    for (int r = 0; r < 8; r++) { mI[r] = -3.0e38f; lI[r] = 0.f; }
    const float scale = 0.125f;  // 1/sqrt(64)

    for (int kb = 0; kb < SEQ; kb += 32) {
        __syncthreads();  // previous iteration's LDS reads complete (WAR)
        // Stage K tile (32x64) and V^T tile (64x32): 256 chunks each, 2/thread
#pragma unroll
        for (int i = 0; i < 2; i++) {
            int idx = t + i * 128;
            int kr = idx >> 3, kc = (idx & 7) * 8;
            cp16(&Ks[kr][kc], Kb + ((size_t)h * SEQ + kb + kr) * HD + kc);
            int vr = idx >> 2, vc = (idx & 3) * 8;
            cp16(&Vs[vr][vc], Vt + ((size_t)h * HD + vr) * SEQ + kb + vc);
        }
        cp_commit();
        __syncthreads();

        // scores: S[16 q][32 keys] as two 16x16 WMMA accumulators
        v8f sc[2];
#pragma unroll
        for (int nt = 0; nt < 2; nt++) {
            v8f z = {};
#pragma unroll
            for (int ks = 0; ks < 2; ks++) {
                const __bf16* p = &Ks[nt * 16 + ln][ks * 32 + half * 8];
                z = wmma_bf16(aq[ks], frag16(p, p + 16), z);
            }
            sc[nt] = z * scale;
        }
        // online softmax per row (rows stay within a 16-lane half)
#pragma unroll
        for (int r = 0; r < 8; r++) {
            float v = fmaxf(sc[0][r], sc[1][r]);
#pragma unroll
            for (int off = 8; off >= 1; off >>= 1) v = fmaxf(v, __shfl_xor(v, off, 32));
            float mn = fmaxf(mI[r], v);
            float sf = __expf(mI[r] - mn);
            mI[r] = mn;
            lI[r] *= sf;
            acc[0][r] *= sf; acc[1][r] *= sf; acc[2][r] *= sf; acc[3][r] *= sf;
            float p0 = __expf(sc[0][r] - mn);
            float p1 = __expf(sc[1][r] - mn);
            sc[0][r] = p0; sc[1][r] = p1;
            float rs = p0 + p1;
#pragma unroll
            for (int off = 8; off >= 1; off >>= 1) rs += __shfl_xor(rs, off, 32);
            lI[r] += rs;
        }
        // C-layout -> A-layout for P via per-wave LDS tile (DS in-order/wave)
#pragma unroll
        for (int r = 0; r < 8; r++) {
            pbuf[wid][r + half * 8][ln]      = (__bf16)sc[0][r];
            pbuf[wid][r + half * 8][16 + ln] = (__bf16)sc[1][r];
        }
        const __bf16* Pp = &pbuf[wid][ln][half * 8];
        bf16x16 ap = frag16(Pp, Pp + 16);
        // P @ V : B fragments from staged V^T tile
#pragma unroll
        for (int ni = 0; ni < 4; ni++) {
            const __bf16* Vp = &Vs[ni * 16 + ln][half * 8];
            acc[ni] = wmma_bf16(ap, frag16(Vp, Vp + 16), acc[ni]);
        }
    }
    // normalize + store bf16 row-major [s][h*64+d] (A-operand for O-proj GEMM)
#pragma unroll
    for (int ni = 0; ni < 4; ni++) {
        int col = h * HD + ni * 16 + ln;
#pragma unroll
        for (int r = 0; r < 8; r++) {
            int row = q0 + r + half * 8;
            Ob[(size_t)row * DMODEL + col] = (__bf16)(acc[ni][r] / lI[r]);
        }
    }
}

// ---------------------------------------------------------------------------
// Host launcher
// ---------------------------------------------------------------------------
extern "C" void kernel_launch(void* const* d_in, const int* in_sizes, int n_in,
                              void* d_out, int out_size, void* d_ws, size_t ws_size,
                              hipStream_t stream) {
    (void)in_sizes; (void)n_in; (void)out_size; (void)ws_size;
    const float* hs  = (const float*)d_in[0];
    const float* cs  = (const float*)d_in[1];
    const float* sn  = (const float*)d_in[2];
    const float* wqf = (const float*)d_in[3];
    const float* bq  = (const float*)d_in[4];
    const float* wkf = (const float*)d_in[5];
    const float* bk  = (const float*)d_in[6];
    const float* wvf = (const float*)d_in[7];
    const float* bv  = (const float*)d_in[8];
    const float* nqw = (const float*)d_in[9];
    const float* nkw = (const float*)d_in[10];
    const float* wof = (const float*)d_in[11];
    const float* bo  = (const float*)d_in[12];
    float* out = (float*)d_out;

    char* p = (char*)d_ws;
    auto alloc = [&](size_t bytes) -> void* {
        void* r = (void*)p;
        p += (bytes + 255) & ~(size_t)255;
        return r;
    };
    const size_t SD = (size_t)SEQ * DMODEL;
    __bf16* Xb  = (__bf16*)alloc(SD * 2);
    __bf16* Wqt = (__bf16*)alloc(SD * 2);
    __bf16* Wkt = (__bf16*)alloc(SD * 2);
    __bf16* Wvt = (__bf16*)alloc(SD * 2);
    __bf16* Wot = (__bf16*)alloc(SD * 2);
    float*  Qf  = (float*)alloc(SD * 4);
    float*  Kf  = (float*)alloc(SD * 4);
    float*  Vf  = (float*)alloc(SD * 4);
    __bf16* Qb  = (__bf16*)alloc(SD * 2);
    __bf16* Kbb = (__bf16*)alloc(SD * 2);
    __bf16* Vt  = (__bf16*)alloc(SD * 2);
    __bf16* Ab  = (__bf16*)alloc(SD * 2);

    cvt_bf16_kernel<<<(int)(SD / (256 * 8)), 256, 0, stream>>>(hs, Xb);

    dim3 tb(32, 8), tg(DMODEL / 32, DMODEL / 32);
    transpose_w_kernel<<<tg, tb, 0, stream>>>(wqf, Wqt);
    transpose_w_kernel<<<tg, tb, 0, stream>>>(wkf, Wkt);
    transpose_w_kernel<<<tg, tb, 0, stream>>>(wvf, Wvt);
    transpose_w_kernel<<<tg, tb, 0, stream>>>(wof, Wot);

    dim3 gg(DMODEL / BN, SEQ / BM);
    gemm_bf16_kernel<<<gg, 256, 0, stream>>>(Xb, Wqt, bq, Qf);
    gemm_bf16_kernel<<<gg, 256, 0, stream>>>(Xb, Wkt, bk, Kf);
    gemm_bf16_kernel<<<gg, 256, 0, stream>>>(Xb, Wvt, bv, Vf);

    norm_rope_pack_kernel<<<SEQ, 256, 0, stream>>>(Qf, Kf, Vf, nqw, nkw, cs, sn,
                                                   Qb, Kbb, Vt);

    attn_kernel<<<dim3(SEQ / 64, NHEADS), 128, 0, stream>>>(Qb, Kbb, Vt, Ab);

    gemm_bf16_kernel<<<gg, 256, 0, stream>>>(Ab, Wot, bo, out);
}
